// GATNetwork_12352325943710
// MI455X (gfx1250) — matrix operand order repaired and checked by reading, not compiled, not run on hardware
//
#include <hip/hip_runtime.h>
#include <hip/hip_bf16.h>
#include <math.h>

typedef __attribute__((ext_vector_type(16))) _Float16 v16h;
typedef __attribute__((ext_vector_type(8)))  _Float16 v8h;
typedef __attribute__((ext_vector_type(8)))  float    v8f;

#define NN   384
#define DIN  512
#define HH   8
#define FF   64
#define HF   512   // H*F
#define GW   1024  // width of fused [g_l | g_r]
#define SLOPE 0.2f

// One k-step's operand fragments for a 16x64 strip (A + 4 B tiles).
struct FragSet {
    v8h alo, ahi;
    v8h blo[4], bhi[4];
};

// ---------------------------------------------------------------------------
// Core: one wave computes a 16x64 f32 strip (4 adjacent 16x16 WMMA tiles),
// software-pipelined depth-2: fragments for k-step n+1 are loaded into a
// second buffer while k-step n's WMMAs execute, so per-wave loads overlap
// matrix ops instead of serializing behind s_wait_loadcnt 0.
// A:  row-major 16xK f16 (lda elements per row)
// BT: column-major B, i.e. BT[c][k] -> every B-fragment is two runs of 8 f16
//     per lane (two b128 loads, no packing VALU).
// Fragment layouts per CDNA5 ISA 7.12.2 (wave32):
//   A 16x32: lane L row=(L&15), K = koff..koff+7 (v0..3), 16+koff.. (v4..7),
//            koff = 8*(L>=16)
//   B 32x16: lane L col=(L&15), K = koff..koff+7 (v0..3), 16+koff.. (v4..7)
//   C/D f32: lane L col=(L&15), VGPR r -> row = r + 8*(L>=16)
// ---------------------------------------------------------------------------
template <int K>
__device__ __forceinline__ void wmma_strip_x4(const _Float16* __restrict__ Arow, int lda,
                                              const _Float16* __restrict__ BT, int ldk,
                                              int cbase, int lane, v8f acc[4]) {
    static_assert(K % 64 == 0, "pipeline assumes K multiple of 64");
    const int row_a = lane & 15;
    const int koff  = (lane >> 4) * 8;
    const int krow0 = (lane >> 4) * 16;
    const int col_b = lane & 15;
    const _Float16* ap0 = Arow + (size_t)row_a * lda + koff;
    const _Float16* bp0 = BT + (size_t)(cbase + col_b) * ldk + koff;

    auto load_frag = [&](int k0, FragSet& f) {
        const _Float16* ap = ap0 + k0;
        f.alo = *(const v8h*)(ap);
        f.ahi = *(const v8h*)(ap + 16);
        __builtin_prefetch(ap + 64, 0, 1);
#pragma unroll
        for (int t = 0; t < 4; ++t) {
            const _Float16* bp = bp0 + (size_t)(t * 16) * ldk + k0;
            f.blo[t] = *(const v8h*)(bp);
            f.bhi[t] = *(const v8h*)(bp + 16);
        }
    };
    auto do_wmma = [&](const FragSet& f) {
        v16h af = __builtin_shufflevector(f.alo, f.ahi,
                                          0, 1, 2, 3, 4, 5, 6, 7,
                                          8, 9, 10, 11, 12, 13, 14, 15);
#pragma unroll
        for (int t = 0; t < 4; ++t) {
            v16h bf = __builtin_shufflevector(f.blo[t], f.bhi[t],
                                              0, 1, 2, 3, 4, 5, 6, 7,
                                              8, 9, 10, 11, 12, 13, 14, 15);
            acc[t] = __builtin_amdgcn_wmma_f32_16x16x32_f16(
                /*neg_a=*/false, af, /*neg_b=*/false, bf,
                /*c_mod=*/(short)0, acc[t], /*reuse_a=*/false, /*reuse_b=*/false);
        }
    };

    FragSet buf0, buf1;
    load_frag(0, buf0);
    for (int k0 = 0; k0 < K - 64; k0 += 64) {
        load_frag(k0 + 32, buf1);   // overlaps buf0's WMMAs
        do_wmma(buf0);
        load_frag(k0 + 64, buf0);   // overlaps buf1's WMMAs
        do_wmma(buf1);
    }
    load_frag(K - 32, buf1);
    do_wmma(buf0);
    do_wmma(buf1);
}

// ---------------------------------------------------------------------------
// fp32 -> f16 elementwise cast
// ---------------------------------------------------------------------------
__global__ void cvt_f32_to_f16(const float* __restrict__ in,
                               _Float16* __restrict__ out, int n) {
    int i = blockIdx.x * blockDim.x + threadIdx.x;
    if (i < n) out[i] = (_Float16)in[i];
}

// Fuse W_l (512xHF) and W_r (512xHF) and TRANSPOSE into WhT[c][k] (1024x512) f16
__global__ void pack_w(const float* __restrict__ Wl, const float* __restrict__ Wr,
                       _Float16* __restrict__ WhT) {
    int i = blockIdx.x * blockDim.x + threadIdx.x;
    if (i >= GW * DIN) return;
    int c = i / DIN;  // output column 0..1023
    int k = i % DIN;  // input dim
    float v = (c < HF) ? Wl[k * HF + c] : Wr[k * HF + (c - HF)];
    WhT[i] = (_Float16)v;
}

// ---------------------------------------------------------------------------
// g[384 x 1024] = xh[384 x 512] @ Wh[512 x 1024]   (B given transposed: WhT)
// g_l = cols 0..511, g_r = cols 512..1023.  Also writes g_r transposed as f16
// (gr16T[c][j], c=0..511) for the aggregation GEMM's B operand.
// grid(24, 2), block(32, 8): wave = one 16x64 strip (colGroup 0..15).
// ---------------------------------------------------------------------------
__global__ void gemm_g_kernel(const _Float16* __restrict__ xh,
                              const _Float16* __restrict__ WhT,
                              float* __restrict__ g,
                              _Float16* __restrict__ gr16T) {
    const int lane     = threadIdx.x;
    const int wave     = threadIdx.y;
    const int mTile    = blockIdx.x;
    const int colGroup = blockIdx.y * 8 + wave;  // 0..15, 64 cols each
    v8f acc[4] = {};
    wmma_strip_x4<DIN>(xh + (size_t)mTile * 16 * DIN, DIN,
                       WhT, DIN, colGroup * 64, lane, acc);
    const int rbase = (lane >> 4) * 8;
    const int col   = lane & 15;
#pragma unroll
    for (int t = 0; t < 4; ++t) {
        const int c = colGroup * 64 + t * 16 + col;
#pragma unroll
        for (int r = 0; r < 8; ++r) {
            const int row = mTile * 16 + rbase + r;
            g[(size_t)row * GW + c] = acc[t][r];
        }
    }
    if (colGroup >= 8) {  // uniform: this strip is entirely g_r
#pragma unroll
        for (int t = 0; t < 4; ++t) {
            const int c = colGroup * 64 + t * 16 + col;  // >= 512
            v8h hv;
#pragma unroll
            for (int r = 0; r < 8; ++r) hv[r] = (_Float16)acc[t][r];
            // 8 consecutive rows -> one aligned 16B store in transposed layout
            *(v8h*)(gr16T + (size_t)(c - HF) * NN + mTile * 16 + rbase) = hv;
        }
    }
}

// ---------------------------------------------------------------------------
// Scores + masked softmax for one (i, h): e[j] = a . LeakyReLU(g_r[i]+g_l[j])
// grid(384, 8), block(128): thread t owns j = t, t+128, t+256.
// ---------------------------------------------------------------------------
__global__ void scores_softmax(const float* __restrict__ g,
                               const int* __restrict__ adj,
                               const float* __restrict__ avec,
                               _Float16* __restrict__ att16) {
    const int i   = blockIdx.x;
    const int h   = blockIdx.y;
    const int tid = threadIdx.x;
    __shared__ float grf[FF];
    __shared__ float as[FF];
    __shared__ float red[128];
    if (tid < FF) {
        grf[tid] = g[(size_t)i * GW + HF + h * FF + tid];  // g_r[i,h,:]
        as[tid]  = avec[tid];
    }
    __syncthreads();

    float e[3];
    float lmax = -3.0e38f;
#pragma unroll
    for (int jj = 0; jj < 3; ++jj) {
        const int j = tid + jj * 128;
        const float* gl = g + (size_t)j * GW + h * FF;     // g_l[j,h,:]
        float d = 0.f;
#pragma unroll
        for (int f = 0; f < FF; f += 4) {
            float4 v = *(const float4*)(gl + f);
            float z0 = v.x + grf[f + 0];
            float z1 = v.y + grf[f + 1];
            float z2 = v.z + grf[f + 2];
            float z3 = v.w + grf[f + 3];
            z0 = z0 > 0.f ? z0 : SLOPE * z0;
            z1 = z1 > 0.f ? z1 : SLOPE * z1;
            z2 = z2 > 0.f ? z2 : SLOPE * z2;
            z3 = z3 > 0.f ? z3 : SLOPE * z3;
            d = fmaf(z0, as[f + 0], d);
            d = fmaf(z1, as[f + 1], d);
            d = fmaf(z2, as[f + 2], d);
            d = fmaf(z3, as[f + 3], d);
        }
        const bool m = (adj[(size_t)i * NN + j] != 0) || (j == i);  // self-loop
        e[jj] = m ? d : -3.0e38f;
        lmax  = fmaxf(lmax, e[jj]);
    }
    // block max
    red[tid] = lmax; __syncthreads();
    for (int s = 64; s > 0; s >>= 1) {
        if (tid < s) red[tid] = fmaxf(red[tid], red[tid + s]);
        __syncthreads();
    }
    const float rmax = red[0];
    __syncthreads();
    float p[3];
    float lsum = 0.f;
#pragma unroll
    for (int jj = 0; jj < 3; ++jj) { p[jj] = __expf(e[jj] - rmax); lsum += p[jj]; }
    red[tid] = lsum; __syncthreads();
    for (int s = 64; s > 0; s >>= 1) {
        if (tid < s) red[tid] += red[tid + s];
        __syncthreads();
    }
    const float inv = 1.0f / red[0];
    _Float16* dst = att16 + (size_t)h * NN * NN + (size_t)i * NN;
#pragma unroll
    for (int jj = 0; jj < 3; ++jj) dst[tid + jj * 128] = (_Float16)(p[jj] * inv);
}

// ---------------------------------------------------------------------------
// Per-head aggregation: out_h[384x64] = att_h[384x384] @ g_r_h[384x64],
// fused activation (ELU for layer 0) + residual, fp32 out.
// grid(24, 1), block(32, 8): wave w = head w, one 16x64 strip per wave.
// ---------------------------------------------------------------------------
__global__ void gemm_agg_kernel(const _Float16* __restrict__ att16,
                                const _Float16* __restrict__ gr16T,
                                const float* __restrict__ xin,
                                float* __restrict__ xout,
                                int do_elu) {
    const int lane  = threadIdx.x;
    const int h     = threadIdx.y;   // head == colGroup
    const int mTile = blockIdx.x;
    v8f acc[4] = {};
    wmma_strip_x4<NN>(att16 + (size_t)h * NN * NN + (size_t)mTile * 16 * NN, NN,
                      gr16T, NN, h * FF, lane, acc);
    const int rbase = (lane >> 4) * 8;
    const int col   = lane & 15;
    if (do_elu) {
#pragma unroll
        for (int t = 0; t < 4; ++t) {
            const int c = h * FF + t * 16 + col;
#pragma unroll
            for (int r = 0; r < 8; ++r) {
                const int row = mTile * 16 + rbase + r;
                float v = acc[t][r];
                v = v > 0.f ? v : (__expf(v) - 1.0f);
                xout[(size_t)row * HF + c] = xin[(size_t)row * HF + c] + v;
            }
        }
    } else {
#pragma unroll
        for (int t = 0; t < 4; ++t) {
            const int c = h * FF + t * 16 + col;
#pragma unroll
            for (int r = 0; r < 8; ++r) {
                const int row = mTile * 16 + rbase + r;
                xout[(size_t)row * HF + c] = xin[(size_t)row * HF + c] + acc[t][r];
            }
        }
    }
}

// ---------------------------------------------------------------------------
// Host launcher
// ---------------------------------------------------------------------------
extern "C" void kernel_launch(void* const* d_in, const int* in_sizes, int n_in,
                              void* d_out, int out_size, void* d_ws, size_t ws_size,
                              hipStream_t stream) {
    (void)in_sizes; (void)n_in; (void)out_size; (void)ws_size;
    const float* x   = (const float*)d_in[0];
    const int*   adj = (const int*)d_in[1];
    const float* Wl0 = (const float*)d_in[2];
    const float* Wr0 = (const float*)d_in[3];
    const float* a0  = (const float*)d_in[4];
    const float* Wl1 = (const float*)d_in[5];
    const float* Wr1 = (const float*)d_in[6];
    const float* a1  = (const float*)d_in[7];
    float* out = (float*)d_out;

    // scratch carve-out (256B aligned slices)
    char* p = (char*)d_ws;
    auto carve = [&](size_t bytes) -> void* {
        void* r = (void*)p;
        p += (bytes + 255) & ~(size_t)255;
        return r;
    };
    _Float16* xh     = (_Float16*)carve((size_t)NN * DIN * 2);     // layer input f16
    _Float16* WhT0   = (_Float16*)carve((size_t)GW * DIN * 2);     // fused W^T f16 (layer 0)
    _Float16* WhT1   = (_Float16*)carve((size_t)GW * DIN * 2);     // fused W^T f16 (layer 1)
    float*    g      = (float*)   carve((size_t)NN * GW * 4);      // [g_l | g_r] fp32
    _Float16* gr16T  = (_Float16*)carve((size_t)HF * NN * 2);      // g_r^T f16
    _Float16* att16  = (_Float16*)carve((size_t)HH * NN * NN * 2); // attention f16
    float*    x1     = (float*)   carve((size_t)NN * HF * 4);      // layer-0 output fp32

    const dim3 gemmBlk(32, 8);

    // one-time weight packing (fused + transposed)
    pack_w<<<(GW * DIN + 255) / 256, 256, 0, stream>>>(Wl0, Wr0, WhT0);
    pack_w<<<(GW * DIN + 255) / 256, 256, 0, stream>>>(Wl1, Wr1, WhT1);

    // ----- layer 0: GATv2 -> ELU -> residual -----
    cvt_f32_to_f16<<<(NN * DIN + 255) / 256, 256, 0, stream>>>(x, xh, NN * DIN);
    gemm_g_kernel<<<dim3(NN / 16, 2), gemmBlk, 0, stream>>>(xh, WhT0, g, gr16T);
    scores_softmax<<<dim3(NN, HH), 128, 0, stream>>>(g, adj, a0, att16);
    gemm_agg_kernel<<<dim3(NN / 16, 1), gemmBlk, 0, stream>>>(att16, gr16T, x, x1, 1);

    // ----- layer 1: GATv2 -> Identity -> residual -----
    cvt_f32_to_f16<<<(NN * HF + 255) / 256, 256, 0, stream>>>(x1, xh, NN * HF);
    gemm_g_kernel<<<dim3(NN / 16, 2), gemmBlk, 0, stream>>>(xh, WhT1, g, gr16T);
    scores_softmax<<<dim3(NN, HH), 128, 0, stream>>>(g, adj, a1, att16);
    gemm_agg_kernel<<<dim3(NN / 16, 1), gemmBlk, 0, stream>>>(att16, gr16T, x1, out, 0);
}
